// MaskedSelfAttentionLayer_23141283790865
// MI455X (gfx1250) — matrix operand order repaired and checked by reading, not compiled, not run on hardware
//
#include <hip/hip_runtime.h>

typedef __attribute__((ext_vector_type(16))) _Float16 v16h;
typedef __attribute__((ext_vector_type(8)))  _Float16 v8h;
typedef __attribute__((ext_vector_type(8)))  float    v8f;

#define NBATCH 4
#define NSEQ   4096
#define DIN    512
#define DOUT   64
#define BN     (NBATCH * NSEQ)

// LDS tile strides (padded to dodge bank conflicts)
#define KLS 72   // K tile row stride in f16 (64 data + 8 pad)
#define VLS 40   // V tile row stride in f16 (32 data + 8 pad)

static __device__ inline v8f v8f_zero() {
  v8f r;
#pragma unroll
  for (int i = 0; i < 8; ++i) r[i] = 0.0f;
  return r;
}

// D = A(16x32 f16) * B(32x16 f16) + C(16x16 f32)
static __device__ inline v8f wmma16(v16h a, v16h b, v8f c) {
  return __builtin_amdgcn_wmma_f32_16x16x32_f16(false, a, false, b, (short)0, c,
                                                false, false);
}

// A-operand loader (f16 source, global or LDS). p = &row[dchunk + 16*half].
// lanes 0-15: K = 0..15 ; lanes 16-31: K = 16..31.
static __device__ inline v16h load_a16(const _Float16* p) {
  v8h lo = *(const v8h*)(p);
  v8h hi = *(const v8h*)(p + 8);
  v16h r;
#pragma unroll
  for (int i = 0; i < 8; ++i) { r[i] = lo[i]; r[i + 8] = hi[i]; }
  return r;
}

// B-operand loader from two 8-f16 chunks at p and p+16 (16B-aligned is enough).
static __device__ inline v16h load_b16(const _Float16* p) {
  v8h lo = *(const v8h*)(p);
  v8h hi = *(const v8h*)(p + 16);
  v16h r;
#pragma unroll
  for (int i = 0; i < 8; ++i) { r[i] = lo[i]; r[i + 8] = hi[i]; }
  return r;
}

// B-operand loader from f32 row-major W [64, 512]: B = W^T tile.
static __device__ inline v16h load_b_w(const float* W, int t, int kb, int l16, int half) {
  const float* p = W + (size_t)(t * 16 + l16) * DIN + kb + half * 16;
  v16h r;
#pragma unroll
  for (int i = 0; i < 16; i += 4) {
    float4 f = *(const float4*)(p + i);
    r[i] = (_Float16)f.x; r[i + 1] = (_Float16)f.y;
    r[i + 2] = (_Float16)f.z; r[i + 3] = (_Float16)f.w;
  }
  return r;
}

// ---- CDNA5 async copy: 16B global -> LDS, tracked by ASYNCcnt ----
static __device__ inline void async_b128(const _Float16* g, const _Float16* l) {
  const unsigned long long ga = (unsigned long long)(const void*)g;
  const unsigned la = (unsigned)(size_t)l;  // LDS offset = flat addr low 32 bits
  asm volatile("global_load_async_to_lds_b128 %0, %1, off"
               :: "v"(la), "v"(ga) : "memory");
}
static __device__ inline void wait_async_le4() {
  asm volatile("s_wait_asynccnt 0x4" ::: "memory");
}
static __device__ inline void wait_async_le0() {
  asm volatile("s_wait_asynccnt 0x0" ::: "memory");
}

// Cooperative copy of one 32-key block: K tile [32][64] f16, V^T tile [64][32] f16.
// 128 threads x (2 K-slices + 2 V-slices) of 16B each; ASYNCcnt += 4 per thread.
static __device__ inline void issue_copy(const _Float16* Kg, const _Float16* Vg,
                                         _Float16* kbuf, _Float16* vbuf,
                                         int jb, int tid) {
#pragma unroll
  for (int s = 0; s < 2; ++s) {
    const int sid = tid + s * 128;
    const int kr = sid >> 3, kc = sid & 7;   // K: 32 rows x 8 chunks
    async_b128(Kg + (size_t)(jb + kr) * DOUT + kc * 8, kbuf + kr * KLS + kc * 8);
    const int vr = sid >> 2, vc = sid & 3;   // V: 64 rows x 4 chunks
    async_b128(Vg + (size_t)vr * NSEQ + jb + vc * 8, vbuf + vr * VLS + vc * 8);
  }
}

// ---------------------------------------------------------------------------
// Kernel 1: QKV projection.  x [16384, 512] f32 row-major.
// Outputs: Qh [16384,64] f16 (pre-scaled by 1/8), Kh [16384,64] f16,
//          VTh [4][64][4096] f16 (V transposed per batch).
// ---------------------------------------------------------------------------
__global__ __launch_bounds__(256)
void qkv_proj_kernel(const float* __restrict__ x,
                     const float* __restrict__ Wk,
                     const float* __restrict__ Wq,
                     const float* __restrict__ Wv,
                     _Float16* __restrict__ Qh,
                     _Float16* __restrict__ Kh,
                     _Float16* __restrict__ VTh) {
  const int wave = threadIdx.x >> 5;
  const int lane = threadIdx.x & 31;
  const int half = lane >> 4;
  const int l16  = lane & 15;
  const int m0   = (blockIdx.x * 8 + wave) * 16;

  v8f aK[4], aQ[4], aV[4];
#pragma unroll
  for (int t = 0; t < 4; ++t) { aK[t] = v8f_zero(); aQ[t] = v8f_zero(); aV[t] = v8f_zero(); }

  const float* xrow = x + (size_t)(m0 + l16) * DIN + half * 16;

  for (int kb = 0; kb < DIN; kb += 32) {
    v16h a;
    {
      const float* p = xrow + kb;
#pragma unroll
      for (int i = 0; i < 8; i += 4) {
        float4 c0 = *(const float4*)(p + i);
        float4 c1 = *(const float4*)(p + 8 + i);
        a[i] = (_Float16)c0.x; a[i + 1] = (_Float16)c0.y;
        a[i + 2] = (_Float16)c0.z; a[i + 3] = (_Float16)c0.w;
        a[i + 8] = (_Float16)c1.x; a[i + 9] = (_Float16)c1.y;
        a[i + 10] = (_Float16)c1.z; a[i + 11] = (_Float16)c1.w;
      }
    }
#pragma unroll
    for (int t = 0; t < 4; ++t) {
      aK[t] = wmma16(a, load_b_w(Wk, t, kb, l16, half), aK[t]);
      aQ[t] = wmma16(a, load_b_w(Wq, t, kb, l16, half), aQ[t]);
      aV[t] = wmma16(a, load_b_w(Wv, t, kb, l16, half), aV[t]);
    }
  }

  const int b    = m0 >> 12;
  const int mloc = m0 & (NSEQ - 1);
#pragma unroll
  for (int t = 0; t < 4; ++t) {
    const int col = t * 16 + l16;
#pragma unroll
    for (int r = 0; r < 8; ++r) {
      const int row = m0 + r + 8 * half;
      Kh[(size_t)row * DOUT + col] = (_Float16)aK[t][r];
      Qh[(size_t)row * DOUT + col] = (_Float16)(aQ[t][r] * 0.125f);
      VTh[((size_t)b * DOUT + col) * NSEQ + (mloc + r + 8 * half)] = (_Float16)aV[t][r];
    }
  }
}

// ---------------------------------------------------------------------------
// Kernel 2: causal flash attention.
// 4 waves / 128 threads per block; each wave owns one 16-query tile.
// K/V^T 32-key blocks are double-buffered in LDS via async global->LDS copies.
// S^T = K * Q^T so P lands directly in WMMA A-operand layout.
// ---------------------------------------------------------------------------
__global__ __launch_bounds__(128)
void attn_kernel(const _Float16* __restrict__ Qh,
                 const _Float16* __restrict__ Kh,
                 const _Float16* __restrict__ VTh,
                 float* __restrict__ out) {
  __shared__ _Float16 Kt[2][32 * KLS];
  __shared__ _Float16 Vt[2][64 * VLS];

  const int tid  = threadIdx.x;
  const int wave = tid >> 5;
  const int lane = tid & 31;
  const int half = lane >> 4;
  const int l16  = lane & 15;

  const int nqb = NSEQ / 64;                 // 64-query blocks per batch
  const int b   = blockIdx.x / nqb;
  const int q64 = (blockIdx.x % nqb) * 64;   // workgroup query base
  const int q0  = q64 + wave * 16;           // this wave's query tile

  const _Float16* Kg = Kh + (size_t)b * NSEQ * DOUT;
  const _Float16* Vg = VTh + (size_t)b * DOUT * NSEQ;

  // Q^T as B operand: lane holds query n = l16, d contiguous.
  const _Float16* qrow = Qh + (size_t)(b * NSEQ + q0 + l16) * DOUT;
  const v16h qb0 = *(const v16h*)(qrow + half * 16);
  const v16h qb1 = *(const v16h*)(qrow + 32 + half * 16);

  v8f o[4];
#pragma unroll
  for (int t = 0; t < 4; ++t) o[t] = v8f_zero();
  float m_run = -1.0e30f;
  float l_run = 0.0f;

  const int nj_max = (q64 + 64 + 31) >> 5;   // blocks the workgroup must stream
  const int nj_w   = (q0 + 47) >> 5;         // blocks this wave computes

  issue_copy(Kg, Vg, &Kt[0][0], &Vt[0][0], 0, tid);

  for (int j = 0; j < nj_max; ++j) {
    const int buf = j & 1;
    if (j + 1 < nj_max) {
      issue_copy(Kg, Vg, &Kt[buf ^ 1][0], &Vt[buf ^ 1][0], (j + 1) * 32, tid);
      wait_async_le4();   // own slices of block j have landed
    } else {
      wait_async_le0();
    }
    __syncthreads();      // all waves' slices of block j visible

    if (j < nj_w) {       // wave-uniform guard: EXEC stays all-ones inside
      const int jb = j * 32;

      // ---- S^T tiles from LDS K ----
      const _Float16* kp = &Kt[buf][0] + (size_t)l16 * KLS + half * 16;
      v8f st_lo = v8f_zero();
      st_lo = wmma16(load_a16(kp),      qb0, st_lo);
      st_lo = wmma16(load_a16(kp + 32), qb1, st_lo);
      const _Float16* kp2 = kp + 16 * KLS;
      v8f st_hi = v8f_zero();
      st_hi = wmma16(load_a16(kp2),      qb0, st_hi);
      st_hi = wmma16(load_a16(kp2 + 32), qb1, st_hi);

      // ---- causal mask on diagonal-touching blocks ----
      if (jb + 31 > q0) {
        const int q = q0 + l16;
#pragma unroll
        for (int r = 0; r < 8; ++r) {
          const int klo = jb + r + 8 * half;
          if (klo > q)      st_lo[r] = -1.0e30f;
          if (klo + 16 > q) st_hi[r] = -1.0e30f;
        }
      }

      // ---- online softmax (stats per query = per lane) ----
      float mx = -1.0e30f;
#pragma unroll
      for (int r = 0; r < 8; ++r) mx = fmaxf(mx, fmaxf(st_lo[r], st_hi[r]));
      mx = fmaxf(mx, __shfl_xor(mx, 16));
      const float m_new = fmaxf(m_run, mx);
      const float alpha = __expf(m_run - m_new);

      float rs = 0.0f;
      v16h pa;  // P directly in A-operand layout
#pragma unroll
      for (int r = 0; r < 8; ++r) {
        const float plo = __expf(st_lo[r] - m_new);
        const float phi = __expf(st_hi[r] - m_new);
        rs += plo + phi;
        pa[r]     = (_Float16)plo;
        pa[r + 8] = (_Float16)phi;
      }
      rs += __shfl_xor(rs, 16);
      l_run = l_run * alpha + rs;
      m_run = m_new;

      // ---- rescale O (broadcast alpha lane-space -> row-space) ----
      float ar[8];
#pragma unroll
      for (int r = 0; r < 8; ++r) ar[r] = __shfl(alpha, r + 8 * half);
#pragma unroll
      for (int t = 0; t < 4; ++t)
#pragma unroll
        for (int r = 0; r < 8; ++r) o[t][r] *= ar[r];

      // ---- O += P * V from LDS V^T ----
#pragma unroll
      for (int t = 0; t < 4; ++t) {
        const v16h vb = load_b16(&Vt[buf][0] + (size_t)(t * 16 + l16) * VLS + 8 * half);
        o[t] = wmma16(pa, vb, o[t]);
      }
    }

    __syncthreads();      // buffer j free before it is refilled at j+2
  }

  // ---- epilogue: divide by l, store C-layout tiles ----
  float lr[8];
#pragma unroll
  for (int r = 0; r < 8; ++r) lr[r] = __shfl(l_run, r + 8 * half);
  float* obase = out + (size_t)(b * NSEQ + q0) * DOUT;
#pragma unroll
  for (int t = 0; t < 4; ++t)
#pragma unroll
    for (int r = 0; r < 8; ++r)
      obase[(size_t)(r + 8 * half) * DOUT + t * 16 + l16] = o[t][r] / lr[r];
}

// ---------------------------------------------------------------------------
extern "C" void kernel_launch(void* const* d_in, const int* in_sizes, int n_in,
                              void* d_out, int out_size, void* d_ws, size_t ws_size,
                              hipStream_t stream) {
  const float* x  = (const float*)d_in[0];
  const float* Wk = (const float*)d_in[1];
  const float* Wq = (const float*)d_in[2];
  const float* Wv = (const float*)d_in[3];
  float* out = (float*)d_out;

  // Workspace: Qh (2MB) | Kh (2MB) | VTh (2MB), all f16.
  _Float16* Qh  = (_Float16*)d_ws;
  _Float16* Kh  = Qh + (size_t)BN * DOUT;
  _Float16* VTh = Kh + (size_t)BN * DOUT;

  qkv_proj_kernel<<<BN / 16 / 8, 256, 0, stream>>>(x, Wk, Wq, Wv, Qh, Kh, VTh);
  attn_kernel<<<NBATCH * (NSEQ / 64), 128, 0, stream>>>(Qh, Kh, VTh, out);
}
